// SpatialFeatureAlignmentNetwork_37615323579002
// MI455X (gfx1250) — matrix operand order repaired and checked by reading, not compiled, not run on hardware
//
#include <hip/hip_runtime.h>

// MI455X (gfx1250) implementation of the SpatialFeatureAlignmentNetwork reference.
//
// Phases:
//   1) mean_kernel:     feat[8,128] = mean_{H,W} concat(t,rgb)         (bandwidth bound, 256 MB read)
//   2) fc_wmma_kernel:  cp = init_cp + (feat @ fc_w.T + fc_b)          (single wave, v_wmma_f32_16x16x32_f16)
//   3) tps_solve_kernel: solve L * Wm = Y (28x28, 16 RHS) in f64 LDS   (Gauss-Jordan w/ partial pivoting;
//                        K diagonal is zero so pivoting is required)
//   4) tps_sample_kernel: per-pixel TPS grid eval + bilinear border sample (bandwidth bound, ~256 MB)

typedef __attribute__((ext_vector_type(16))) _Float16 v16h;
typedef __attribute__((ext_vector_type(8)))  float    v8f;

// ---------------------------------------------------------------------------
// Phase 1: channel-wise global mean. 1024 blocks = (b, ch) pairs, 256 threads.
// ---------------------------------------------------------------------------
__global__ __launch_bounds__(256) void mean_kernel(const float* __restrict__ t,
                                                   const float* __restrict__ rgb,
                                                   float* __restrict__ feat) {
  const int b  = blockIdx.x >> 7;
  const int ch = blockIdx.x & 127;
  const float* src = (ch < 64) ? (t   + (size_t)(b * 64 + ch)        * 65536)
                               : (rgb + (size_t)(b * 64 + (ch - 64)) * 65536);
  const float4* src4 = (const float4*)src;
  __shared__ float red[256];
  const int tid = threadIdx.x;
  float s = 0.0f;
  for (int i = 0; i < 64; ++i) {
    __builtin_prefetch(&src4[(i + 8) * 256 + tid], 0, 1);  // global_prefetch_b8
    float4 v = src4[i * 256 + tid];
    s += (v.x + v.y) + (v.z + v.w);
  }
  red[tid] = s;
  __syncthreads();
  for (int off = 128; off > 0; off >>= 1) {
    if (tid < off) red[tid] += red[tid + off];
    __syncthreads();
  }
  if (tid == 0) feat[b * 128 + ch] = red[0] * (1.0f / 65536.0f);
}

// ---------------------------------------------------------------------------
// Phase 2: FC via WMMA. One wave (32 threads). D[16x64] = A[16x128] x B[128x64]
// A rows 0..7 = feat batches; B cols 0..49 = fc_w rows (transposed). 16 WMMAs.
// ---------------------------------------------------------------------------
__global__ __launch_bounds__(32) void fc_wmma_kernel(const float* __restrict__ feat,
                                                     const float* __restrict__ fc_w,
                                                     const float* __restrict__ fc_b,
                                                     const float* __restrict__ init_cp,
                                                     float* __restrict__ cp_out) {
  const int l  = threadIdx.x;   // 0..31
  const int m  = l & 15;        // row (A/D) or column (B)
  const int hi = l >> 4;        // lane-half
  v8f acc[4] = {};              // N tiles 0..3 (cols 0..63)

#pragma unroll
  for (int k0 = 0; k0 < 128; k0 += 32) {
    // A fragment: 16-bit A 16x32 layout (ISA 7.12.2):
    // e<8  -> K = 8*hi + e ; e>=8 -> K = 16 + 8*hi + (e-8)
    v16h a;
#pragma unroll
    for (int e = 0; e < 16; ++e) {
      int K = k0 + ((e < 8) ? (8 * hi + e) : (16 + 8 * hi + (e - 8)));
      float fv = feat[(m & 7) * 128 + K];          // clamped index, always in-bounds
      a[e] = (_Float16)((m < 8) ? fv : 0.0f);
    }
#pragma unroll
    for (int tt = 0; tt < 4; ++tt) {
      int n  = 16 * tt + m;                        // output column
      int nc = (n < 50) ? n : 0;                   // clamp for safe load
      // B fragment: 32x16, lane holds column n; K matches A: e<8 -> 8*hi + e ; e>=8 -> 16 + 8*hi + (e-8)
      v16h bf;
#pragma unroll
      for (int e = 0; e < 16; ++e) {
        int K = k0 + ((e < 8) ? (8 * hi + e) : (16 + 8 * hi + (e - 8)));
        float wv = fc_w[nc * 128 + K];
        bf[e] = (_Float16)((n < 50) ? wv : 0.0f);
      }
      acc[tt] = __builtin_amdgcn_wmma_f32_16x16x32_f16(
          /*neg_a=*/false, a, /*neg_b=*/false, bf,
          /*c_mod=*/(short)0, acc[tt], /*reuse_a=*/false, /*reuse_b=*/false);
    }
  }

  // C/D layout: VGPR v, lanes 0..15 hold rows M=v (batches 0..7 all live here).
  if (hi == 0) {
#pragma unroll
    for (int tt = 0; tt < 4; ++tt) {
      int j = 16 * tt + m;                         // output index 0..49
      if (j < 50) {
#pragma unroll
        for (int v = 0; v < 8; ++v) {              // v = batch
          float delta = acc[tt][v] + fc_b[j];
          cp_out[v * 50 + j] = init_cp[j] + delta; // cp flat [B,25,2]
        }
      }
    }
  }
}

// ---------------------------------------------------------------------------
// Phase 3: TPS solve in f64. L[28x28] augmented with 16 RHS columns (8 batches
// x 2 coords). Gauss-Jordan with partial pivoting (K diagonal is zero!).
// ---------------------------------------------------------------------------
__global__ __launch_bounds__(64) void tps_solve_kernel(const float* __restrict__ init_cp,
                                                       const float* __restrict__ cp,
                                                       float* __restrict__ Wm_out) {
  __shared__ double A[28][44];
  __shared__ double P[25][2];
  __shared__ int pivRow;
  const int j = threadIdx.x;

  if (j < 50) P[j >> 1][j & 1] = (double)init_cp[j];
  __syncthreads();

  if (j < 44) {
    for (int i = 0; i < 28; ++i) {
      double v = 0.0;
      if (j < 25) {
        if (i < 25) {
          double dx = P[i][0] - P[j][0];
          double dy = P[i][1] - P[j][1];
          double d2 = dx * dx + dy * dy;
          v = d2 * log(d2 + 1e-6);
        } else if (i == 25) v = 1.0;
        else if (i == 26)   v = P[j][0];
        else                v = P[j][1];
      } else if (j < 28) {
        if (i < 25) v = (j == 25) ? 1.0 : ((j == 26) ? P[i][0] : P[i][1]);
      } else {
        int b = (j - 28) >> 1, c = (j - 28) & 1;
        if (i < 25) v = (double)cp[b * 50 + i * 2 + c];
      }
      A[i][j] = v;
    }
  }
  __syncthreads();

  for (int p = 0; p < 28; ++p) {
    if (j == 0) {
      int best = p; double bm = fabs(A[p][p]);
      for (int r = p + 1; r < 28; ++r) {
        double m2 = fabs(A[r][p]);
        if (m2 > bm) { bm = m2; best = r; }
      }
      pivRow = best;
    }
    __syncthreads();
    if (j < 44 && pivRow != p) {
      double tsw = A[p][j]; A[p][j] = A[pivRow][j]; A[pivRow][j] = tsw;
    }
    __syncthreads();
    double pv = A[p][p];
    __syncthreads();
    if (j < 44) A[p][j] /= pv;
    __syncthreads();
    for (int r = 0; r < 28; ++r) {
      double f = A[r][p];
      __syncthreads();
      if (j < 44 && r != p) A[r][j] -= f * A[p][j];
      __syncthreads();
    }
  }

  if (j < 16) {
    int b = j >> 1, c = j & 1;
    for (int i = 0; i < 28; ++i)
      Wm_out[b * 56 + i * 2 + c] = (float)A[i][28 + j];
  }
}

// ---------------------------------------------------------------------------
// Phase 4: fused TPS grid evaluation + bilinear border-clamped grid sample.
// Block = (b, h); thread = w. Each thread computes its warp coords once,
// then samples all 64 channels (coalesced loads/stores across w).
// ---------------------------------------------------------------------------
__global__ __launch_bounds__(256) void tps_sample_kernel(const float* __restrict__ t,
                                                         const float* __restrict__ Wm,
                                                         const float* __restrict__ init_cp,
                                                         float* __restrict__ out) {
  __shared__ float sW[56];
  __shared__ float sP[50];
  const int b   = blockIdx.x >> 8;
  const int h   = blockIdx.x & 255;
  const int tid = threadIdx.x;
  if (tid < 56)                sW[tid]      = Wm[b * 56 + tid];
  if (tid >= 64 && tid < 114)  sP[tid - 64] = init_cp[tid - 64];
  __syncthreads();

  const int   w  = tid;
  const float gx = -1.0f + (2.0f / 255.0f) * (float)w;   // x varies with width index
  const float gy = -1.0f + (2.0f / 255.0f) * (float)h;

  // g_aug = [U_0..U_24, 1, gx, gy] ; grid = g_aug @ Wm[b]
  float sx = sW[50] + gx * sW[52] + gy * sW[54];
  float sy = sW[51] + gx * sW[53] + gy * sW[55];
#pragma unroll
  for (int n = 0; n < 25; ++n) {
    float dx = gx - sP[2 * n];
    float dy = gy - sP[2 * n + 1];
    float d2 = dx * dx + dy * dy;
    float u  = d2 * logf(d2 + 1e-6f);
    sx += u * sW[2 * n];
    sy += u * sW[2 * n + 1];
  }

  // bilinear, border padding, align_corners=True
  float x = fminf(fmaxf((sx + 1.0f) * 0.5f * 255.0f, 0.0f), 255.0f);
  float y = fminf(fmaxf((sy + 1.0f) * 0.5f * 255.0f, 0.0f), 255.0f);
  float x0f = floorf(x), y0f = floorf(y);
  int x0 = (int)x0f, y0 = (int)y0f;
  int x1 = min(x0 + 1, 255), y1 = min(y0 + 1, 255);
  float wx = x - x0f, wy = y - y0f;

  const float* tb = t   + (size_t)b * 64 * 65536;
  float*       ob = out + (size_t)b * 64 * 65536 + h * 256 + w;
  const int i00 = (y0 << 8) + x0, i01 = (y0 << 8) + x1;
  const int i10 = (y1 << 8) + x0, i11 = (y1 << 8) + x1;

  for (int c = 0; c < 64; ++c) {
    const float* p = tb + (size_t)c * 65536;
    float v00 = p[i00], v01 = p[i01], v10 = p[i10], v11 = p[i11];
    float top = v00 * (1.0f - wx) + v01 * wx;
    float bot = v10 * (1.0f - wx) + v11 * wx;
    ob[(size_t)c * 65536] = top * (1.0f - wy) + bot * wy;
  }
}

// ---------------------------------------------------------------------------
extern "C" void kernel_launch(void* const* d_in, const int* in_sizes, int n_in,
                              void* d_out, int out_size, void* d_ws, size_t ws_size,
                              hipStream_t stream) {
  const float* t       = (const float*)d_in[0];
  const float* rgb     = (const float*)d_in[1];
  const float* fc_w    = (const float*)d_in[2];
  const float* fc_b    = (const float*)d_in[3];
  const float* init_cp = (const float*)d_in[4];

  float* out    = (float*)d_out;
  float* cp_out = out + (size_t)8 * 64 * 256 * 256;     // cp appended after image output

  float* feat = (float*)d_ws;                            // 1024 floats
  float* Wm   = (float*)((char*)d_ws + 4096);            // 8*28*2 floats

  mean_kernel<<<1024, 256, 0, stream>>>(t, rgb, feat);
  fc_wmma_kernel<<<1, 32, 0, stream>>>(feat, fc_w, fc_b, init_cp, cp_out);
  tps_solve_kernel<<<1, 64, 0, stream>>>(init_cp, cp_out, Wm);
  tps_sample_kernel<<<2048, 256, 0, stream>>>(t, Wm, init_cp, out);
}
